// ChartOperator_5531917878036
// MI455X (gfx1250) — matrix operand order repaired and checked by reading, not compiled, not run on hardware
//
#include <hip/hip_runtime.h>

// ---------------------------------------------------------------------------
// Types for CDNA5 WMMA (wave32): V_WMMA_F32_16X16X32_BF16
// ---------------------------------------------------------------------------
typedef __attribute__((ext_vector_type(16))) __bf16 v16bf;
typedef __attribute__((ext_vector_type(8)))  float  v8f;
typedef __attribute__((ext_vector_type(4)))  int    v4i;

union BF16x8 { uint4 u; __bf16 h[8]; };

__device__ __forceinline__ float bf2f(unsigned short u) {
    unsigned x = ((unsigned)u) << 16;
    float f; __builtin_memcpy(&f, &x, 4); return f;
}
__device__ __forceinline__ unsigned short f2bf(float f) {
    __bf16 b = (__bf16)f;                  // hardware RNE f32->bf16
    unsigned short u; __builtin_memcpy(&u, &b, 2); return u;
}

// ---------------------------------------------------------------------------
// CDNA5 async global->LDS path (ASYNCcnt-tracked, no VGPR round trip).
// Probe round 2 confirmed the builtin exists with v4i32 pointer params:
//   (v4i32 __device__ *gsrc, v4i32 __shared__ *ldsdst, imm offset, imm cpol)
// ---------------------------------------------------------------------------
#define AS1 __attribute__((address_space(1)))
#define AS3 __attribute__((address_space(3)))

#if __has_builtin(__builtin_amdgcn_global_load_async_to_lds_b128)
#define HAVE_ASYNC_LDS 1
#else
#define HAVE_ASYNC_LDS 0
#endif

__device__ __forceinline__ void async_copy16(const void* src, void* dst) {
#if HAVE_ASYNC_LDS
    __builtin_amdgcn_global_load_async_to_lds_b128(
        (AS1 v4i*)(AS1 void*)src, (AS3 v4i*)(AS3 void*)dst, 0, 0);
#endif
}

__device__ __forceinline__ void wait_asynccnt0() {
#if __has_builtin(__builtin_amdgcn_s_wait_asynccnt)
    __builtin_amdgcn_s_wait_asynccnt(0);
#else
    asm volatile("s_wait_asynccnt 0" ::: "memory");
#endif
}

// ---------------------------------------------------------------------------
// Problem constants
// ---------------------------------------------------------------------------
#define BB   4
#define NN   4096
#define PD   1024
#define DD   16
#define SS   64
#define MROWS (BB*NN)          // 16384
#define TWOALL 2048            // 2*D*S

// ===========================================================================
// Kernel 0: prepack weight matrix (f32 row-major K x N) into bf16 fragments
// matching the assumed 16-bit B-matrix (32x16) lane layout:
//   lane L: N = L%16 ; element e (0..15): K = (e&7) + 8*(L/16) + 16*(e>>3)
// Packed index: (((nt*KT + kt)*32 + lane)*16 + e)
// ===========================================================================
__global__ __launch_bounds__(256)
void prepack_B(const float* __restrict__ W, unsigned short* __restrict__ out,
               int K, int Nn) {
    const int KT = K >> 5;
    const long total = (long)K * Nn;
    for (long idx = blockIdx.x * 256 + threadIdx.x; idx < total;
         idx += (long)gridDim.x * 256) {
        int e    = (int)(idx & 15);
        int lane = (int)((idx >> 4) & 31);
        long tile = idx >> 9;
        int kt = (int)(tile % KT);
        int nt = (int)(tile / KT);
        int k = kt * 32 + (e & 7) + 8 * (lane >> 4) + 16 * (e >> 3);
        int n = nt * 16 + (lane & 15);
        out[idx] = f2bf(W[(size_t)k * Nn + n]);
    }
}

// ===========================================================================
// GEMM: Out[M,Nn] = A[M,K] * W + bias   (W prepacked bf16, f32 accumulate)
//   AF32:  A is f32 row-major (converted to bf16 on the fly), else bf16.
//   OUTF32: store f32, else bf16.
// Block: 256 threads (8 waves), block tile 128x128, wave tile 32x64.
// A fragment (16-bit 16x32 layout): lane L holds M=L%16 and two contiguous
// runs of 8 K-values starting at 8*(L/16) and 16+8*(L/16).
// ===========================================================================
template <bool AF32, bool OUTF32>
__global__ __launch_bounds__(256)
void gemm_bf16_wmma(const void* __restrict__ Aptr,
                    const unsigned short* __restrict__ Bp,
                    const float* __restrict__ bias,
                    void* __restrict__ Outp,
                    int M, int K, int Nn) {
    const int tid  = threadIdx.x;
    const int wave = tid >> 5, lane = tid & 31;
    const int h    = lane >> 4, ln16 = lane & 15;
    const int wm   = wave >> 1, wn = wave & 1;
    const int m0   = blockIdx.y * 128 + wm * 32;
    const int n0   = blockIdx.x * 128 + wn * 64;
    const int KT   = K >> 5;

    v8f acc[2][4];
#pragma unroll
    for (int ni = 0; ni < 4; ++ni) {
        float bv = bias[n0 + ni * 16 + ln16];
#pragma unroll
        for (int mi = 0; mi < 2; ++mi)
#pragma unroll
            for (int v = 0; v < 8; ++v) acc[mi][ni][v] = bv;
    }

    for (int kt = 0; kt < KT; ++kt) {
        // L2 prefetch of next k-step tiles (speculative; OOB dropped by HW)
        {
            const int kbn = (kt + 1) * 32 + 8 * h;
            if (AF32)
                __builtin_prefetch((const float*)Aptr + (size_t)(m0 + ln16) * K + kbn, 0, 3);
            else
                __builtin_prefetch((const unsigned short*)Aptr + (size_t)(m0 + ln16) * K + kbn, 0, 3);
            __builtin_prefetch(Bp + ((((size_t)(n0 >> 4)) * KT + kt + 1) * 32 + lane) * 16, 0, 3);
        }

        v16bf a[2];
#pragma unroll
        for (int mi = 0; mi < 2; ++mi) {
            const int mrow = m0 + mi * 16 + ln16;
            const int kb   = kt * 32 + 8 * h;
            if (AF32) {
                const float* ap = (const float*)Aptr + (size_t)mrow * K + kb;
#pragma unroll
                for (int e = 0; e < 8; ++e) a[mi][e]     = (__bf16)ap[e];
#pragma unroll
                for (int e = 0; e < 8; ++e) a[mi][e + 8] = (__bf16)ap[16 + e];
            } else {
                const unsigned short* ap =
                    (const unsigned short*)Aptr + (size_t)mrow * K + kb;
                BF16x8 lo, hi;
                lo.u = *(const uint4*)ap;
                hi.u = *(const uint4*)(ap + 16);
#pragma unroll
                for (int e = 0; e < 8; ++e) { a[mi][e] = lo.h[e]; a[mi][e + 8] = hi.h[e]; }
            }
        }
        v16bf bfr[4];
#pragma unroll
        for (int ni = 0; ni < 4; ++ni) {
            const unsigned short* bp =
                Bp + ((((size_t)(n0 >> 4) + ni) * KT + kt) * 32 + lane) * 16;
            BF16x8 lo, hi;
            lo.u = *(const uint4*)bp;
            hi.u = *(const uint4*)(bp + 8);
#pragma unroll
            for (int e = 0; e < 8; ++e) { bfr[ni][e] = lo.h[e]; bfr[ni][e + 8] = hi.h[e]; }
        }
#pragma unroll
        for (int mi = 0; mi < 2; ++mi)
#pragma unroll
            for (int ni = 0; ni < 4; ++ni)
                acc[mi][ni] = __builtin_amdgcn_wmma_f32_16x16x32_bf16(
                    false, a[mi], false, bfr[ni], (short)0, acc[mi][ni],
                    false, false);
    }

    // C/D layout: lane L: N = L%16; element v: M = v + 8*(L/16)
#pragma unroll
    for (int mi = 0; mi < 2; ++mi)
#pragma unroll
        for (int ni = 0; ni < 4; ++ni) {
            const int col = n0 + ni * 16 + ln16;
#pragma unroll
            for (int v = 0; v < 8; ++v) {
                const int row = m0 + mi * 16 + v + 8 * h;
                if (OUTF32)
                    ((float*)Outp)[(size_t)row * Nn + col] = acc[mi][ni][v];
                else
                    ((unsigned short*)Outp)[(size_t)row * Nn + col] =
                        f2bf(acc[mi][ni][v]);
            }
        }
}

// ===========================================================================
// Chart kernel: one workgroup per (b, chunk c). Builds banded Lg (softmaxed)
// and banded R matrices from LDS-staged P rows, then per s:
//   Q0 = [L0|L1] @ [R0;0],  Q1 = [L0|L1] @ [R1;R0next]
// with one v_wmma_f32_16x16x32_bf16 each, followed by the BWD gather scatter
// into Q (bf16, row-major [16384 x 1024], feature = d*64 + s).
// All banded reads / output scatters are branchless (clamped index + select)
// to avoid exec-mask save/restore around every ds_load / global_store.
// ===========================================================================
#define PITCH 1040   // bf16 units per LDS row (1024 + pad, 16B aligned)

__global__ __launch_bounds__(256)
void chart_kernel(const unsigned short* __restrict__ P,
                  unsigned short* __restrict__ Q) {
    __shared__ unsigned short Lt[16 * PITCH];  // L rows t=0..15 (cols 0..1023)
    __shared__ unsigned short Rt[32 * PITCH];  // R rows r=0..31 (cols 1024..2047)

    const int bc = blockIdx.x;
    const int b  = bc >> 8;
    const int c  = bc & 255;
    const int tid = threadIdx.x;

    // ---- stage tiles: 16 L rows + 32 R rows (current + next chunk, wrapped)
#if HAVE_ASYNC_LDS
    for (int i = tid; i < 16 * 128; i += 256) {
        int t = i >> 7, ch = i & 127;
        async_copy16(P + ((size_t)(b * NN + c * 16 + t)) * TWOALL + ch * 8,
                     Lt + t * PITCH + ch * 8);
    }
    for (int i = tid; i < 32 * 128; i += 256) {
        int t = i >> 7, ch = i & 127;
        int n = (c * 16 + 1 + t) & (NN - 1);           // jnp.roll wrap within b
        async_copy16(P + ((size_t)(b * NN + n)) * TWOALL + 1024 + ch * 8,
                     Rt + t * PITCH + ch * 8);
    }
    wait_asynccnt0();
#else
    for (int i = tid; i < 16 * 128; i += 256) {
        int t = i >> 7, ch = i & 127;
        const uint4 v = *(const uint4*)(P + ((size_t)(b * NN + c * 16 + t)) * TWOALL + ch * 8);
        *(uint4*)(Lt + t * PITCH + ch * 8) = v;
    }
    for (int i = tid; i < 32 * 128; i += 256) {
        int t = i >> 7, ch = i & 127;
        int n = (c * 16 + 1 + t) & (NN - 1);
        const uint4 v = *(const uint4*)(P + ((size_t)(b * NN + n)) * TWOALL + 1024 + ch * 8);
        *(uint4*)(Rt + t * PITCH + ch * 8) = v;
    }
#endif
    __syncthreads();

    const int wave = tid >> 5, lane = tid & 31;
    const int h = lane >> 4;           // half-wave select
    const int t = lane & 15;           // A-matrix row M
    const int u = lane & 15;           // B/D-matrix column N
    const bool lastc = (c == 255);

    for (int it = 0; it < 8; ++it) {
        const int s = wave + 8 * it;   // 0..63

        // ---- A = softmax(Lg) as bf16 fragment (branchless banded reads).
        // element e -> K = tau = e + 16*h ; Lg[t,tau]=L[t, tau-t]
        float vals[16];
        float mx = -3.0e38f;
#pragma unroll
        for (int e = 0; e < 16; ++e) {
            int tau = e + 16 * h;
            int j  = tau - t;
            int jc = min(max(j, 0), 15);                 // clamp: always load
            float raw = bf2f(Lt[t * PITCH + jc * 64 + s]);
            float v = (j == jc) ? raw : -1.0e4f;         // v_cndmask
            vals[e] = v;
            mx = fmaxf(mx, v);
        }
        mx = fmaxf(mx, __shfl_xor(mx, 16, 32));   // combine tau-halves
        float sum = 0.f;
#pragma unroll
        for (int e = 0; e < 16; ++e) { vals[e] = __expf(vals[e] - mx); sum += vals[e]; }
        sum += __shfl_xor(sum, 16, 32);
        const float inv = 1.0f / sum;
        v16bf a;
#pragma unroll
        for (int e = 0; e < 16; ++e) a[e] = (__bf16)(vals[e] * inv);

        // ---- B0 = [R0 ; 0], B1 = [R1 ; R0next]  (branchless banded reads)
        // K row of B: k = e + 16*h.
        //   B0[k][u]: h==0: row e,    j = u-1-e, valid j>=0 ; h==1: zero
        //   B1[k][u]: h==0: row e,    j = 15+u-e, valid j<16
        //             h==1: row 16+e, j = u-1-e,  valid j>=0
        v16bf b0, b1;
#pragma unroll
        for (int e = 0; e < 16; ++e) {
            int j0      = u - 1 - e;
            int j0c     = max(j0, 0);
            float r0raw = bf2f(Rt[e * PITCH + j0c * 64 + s]);
            float f0    = (!h && j0 >= 0) ? r0raw : 0.0f;

            int   row1  = h ? (16 + e) : e;
            int   j1    = h ? (u - 1 - e) : (15 + u - e);
            int   j1c   = min(max(j1, 0), 15);
            bool  ok1   = h ? (j1 >= 0) : (j1 < 16);
            float r1raw = bf2f(Rt[row1 * PITCH + j1c * 64 + s]);
            float f1    = ok1 ? r1raw : 0.0f;

            b0[e] = (__bf16)f0;
            b1[e] = (__bf16)f1;
        }

        v8f zero = {};
        v8f q0 = __builtin_amdgcn_wmma_f32_16x16x32_bf16(
            false, a, false, b0, (short)0, zero, false, false);
        v8f q1 = __builtin_amdgcn_wmma_f32_16x16x32_bf16(
            false, a, false, b1, (short)0, zero, false, false);

        // ---- BWD gather + store: Qout[t, d] = Qcat[t, t+d], d in [0,16)
        // tau<16 -> Q0 (u>=t), tau>=16 -> Q1 (u<t): exactly one per (lane,v)
        // => branchless select of value and offset, unconditional store.
        const size_t rowbase = ((size_t)(b * NN + c * 16)) * (DD * SS);
#pragma unroll
        for (int v = 0; v < 8; ++v) {
            const int  tt    = v + 8 * h;                    // output row M
            const bool fromQ0 = (u >= tt);
            const float q1m  = lastc ? 0.0f : q1[v];         // last-chunk mask
            const float val  = fromQ0 ? q0[v] : q1m;
            const int   d    = fromQ0 ? (u - tt) : (16 + u - tt);
            Q[rowbase + (size_t)tt * (DD * SS) + (size_t)d * 64 + s] = f2bf(val);
        }
    }
}

// ===========================================================================
// Host launcher.  Workspace layout (requires >= 102 MB):
//   [0,   64MB)  P   bf16  16384 x 2048
//   [64,  96MB)  Q   bf16  16384 x 1024
//   [96, 100MB)  W_r packed bf16 (4MB)
//   [100,102MB)  W_w packed bf16 (2MB)
// ===========================================================================
extern "C" void kernel_launch(void* const* d_in, const int* in_sizes, int n_in,
                              void* d_out, int out_size, void* d_ws, size_t ws_size,
                              hipStream_t stream) {
    const float* x   = (const float*)d_in[0];
    const float* W_r = (const float*)d_in[1];
    const float* b_r = (const float*)d_in[2];
    const float* W_w = (const float*)d_in[3];
    const float* b_w = (const float*)d_in[4];

    char* ws = (char*)d_ws;
    unsigned short* Pbuf = (unsigned short*)ws;
    unsigned short* Qbuf = (unsigned short*)(ws + ((size_t)64 << 20));
    unsigned short* WrP  = (unsigned short*)(ws + ((size_t)96 << 20));
    unsigned short* WwP  = (unsigned short*)(ws + ((size_t)100 << 20));

    prepack_B<<<dim3(512), dim3(256), 0, stream>>>(W_r, WrP, PD, TWOALL);
    prepack_B<<<dim3(256), dim3(256), 0, stream>>>(W_w, WwP, PD, PD);

    // GEMM1: P = x @ W_r + b_r   (M=16384, K=1024, N=2048), bf16 out
    gemm_bf16_wmma<true, false><<<dim3(TWOALL / 128, MROWS / 128),
                                  dim3(256), 0, stream>>>(
        x, WrP, b_r, Pbuf, MROWS, PD, TWOALL);

    // Chart core: softmax + banded WMMA matmuls + gathers -> Q (bf16)
    chart_kernel<<<dim3(BB * (NN / DD)), dim3(256), 0, stream>>>(Pbuf, Qbuf);

    // GEMM2: out = Q @ W_w + b_w (M=16384, K=1024, N=1024), f32 out
    gemm_bf16_wmma<false, true><<<dim3(PD / 128, MROWS / 128),
                                  dim3(256), 0, stream>>>(
        Qbuf, WwP, b_w, d_out, MROWS, PD, PD);
}
